// PureLowRankElmanCell_49615462203415
// MI455X (gfx1250) — matrix operand order repaired and checked by reading, not compiled, not run on hardware
//
#include <hip/hip_runtime.h>
#include <hip/hip_bf16.h>
#include <math.h>

// ---------------------------------------------------------------------------
// PureLowRankElman on MI455X (gfx1250, wave32, WMMA bf16 16x16x32)
//   out[t] = h[t+1] * silu(uz[t]),  h[t+1] = tanh(UhVh h[t] + UxVx x[t] + b)
// ---------------------------------------------------------------------------

typedef __bf16 bf16;
typedef __attribute__((ext_vector_type(16))) __bf16 v16bf;
typedef __attribute__((ext_vector_type(8)))  __bf16 v8bf;
typedef __attribute__((ext_vector_type(8)))  float  v8f;

constexpr int kT = 2048;
constexpr int kB = 16;
constexpr int kD = 1024;
constexpr int kR = 64;
constexpr int kBD = kB * kD;          // 16384
constexpr int kW = 65536;             // elements per 64x1024 weight matrix
constexpr int kPadD = kD + 16;        // 1040: keeps 32B alignment of chunks, de-banks rows
constexpr int kPadP = 80;             // padded row for P buffers (f32)

__device__ __forceinline__ bf16 f2bf(float f) {
  unsigned u = __float_as_uint(f);
  u = (u + 0x7FFFu + ((u >> 16) & 1u)) >> 16;   // round-to-nearest-even
  unsigned short s = (unsigned short)u;
  bf16 b;
  __builtin_memcpy(&b, &s, 2);
  return b;
}

// A fragment (16x32 bf16, rows = M): lane holds row (lane&15); chunk0 at
// k = 8*(lane>>4), chunk1 at +16 (each 8 contiguous bf16).
__device__ __forceinline__ v16bf load_a(const bf16* base, int stride, int r,
                                        int kbase, int hi) {
  const bf16* p0 = base + (size_t)r * stride + kbase + hi * 8;
  v8bf c0 = *(const v8bf*)p0;
  v8bf c1 = *(const v8bf*)(p0 + 16);
  v16bf a;
#pragma unroll
  for (int e = 0; e < 8; ++e) { a[e] = c0[e]; a[e + 8] = c1[e]; }
  return a;
}

// B fragment (32x16 bf16, cols = N) from row-major W[n][k]: lane holds column
// n = (lane&15); chunk0 at k = kbase + 8*(lane>>4), chunk1 at +16.
__device__ __forceinline__ v16bf load_b(const bf16* base, int stride, int n,
                                        int kbase, int hi) {
  const bf16* p0 = base + (size_t)n * stride + kbase + hi * 8;
  v8bf c0 = *(const v8bf*)p0;
  v8bf c1 = *(const v8bf*)(p0 + 16);
  v16bf b;
#pragma unroll
  for (int e = 0; e < 8; ++e) { b[e] = c0[e]; b[e + 8] = c1[e]; }
  return b;
}

__device__ __forceinline__ v8f wmma_bf16(v16bf a, v16bf b, v8f c) {
  return __builtin_amdgcn_wmma_f32_16x16x32_bf16(false, a, false, b,
                                                 (short)0, c, false, false);
}

// ---------------------------------------------------------------------------
// Kernel 0: convert the six 64K-element weight matrices f32 -> bf16 into ws.
// ws layout (each kW elems): [Vh, Uh, Vx, Ux, Vz, Uz]
// V* are [R=64][D=1024] row-major (rows n over R, k over D)  -> stage-A B op.
// U* are [D=1024][R=64] row-major (rows n over D, k over R)  -> stage-B B op.
// ---------------------------------------------------------------------------
__global__ __launch_bounds__(256) void cvt_weights(
    const float* __restrict__ Vh, const float* __restrict__ Uh,
    const float* __restrict__ Vx, const float* __restrict__ Ux,
    const float* __restrict__ Vz, const float* __restrict__ Uz,
    bf16* __restrict__ ws) {
  int i = blockIdx.x * 256 + threadIdx.x;
  if (i >= 6 * kW) return;
  int sel = i >> 16;
  int off = i & (kW - 1);
  float v;
  switch (sel) {
    case 0:  v = Vh[off]; break;
    case 1:  v = Uh[off]; break;
    case 2:  v = Vx[off]; break;
    case 3:  v = Ux[off]; break;
    case 4:  v = Vz[off]; break;
    default: v = Uz[off]; break;
  }
  ws[i] = f2bf(v);
}

// ---------------------------------------------------------------------------
// Kernel 1: per-timestep input projections (fully parallel over T).
//   stage A: rx = x_t @ Vx^T, rz = x_t @ Vz^T      ([16 x 64])
//   stage B: ux = rx @ Ux^T + b  -> staged at h[t+1]
//            gate = silu(rz @ Uz^T) -> staged at out[t]
// One block per timestep, 256 threads (8 waves).
// ---------------------------------------------------------------------------
__global__ __launch_bounds__(256) void lr_input_proj(
    const float* __restrict__ x, const float* __restrict__ h0,
    const bf16* __restrict__ wVx, const bf16* __restrict__ wUx,
    const bf16* __restrict__ wVz, const bf16* __restrict__ wUz,
    const float* __restrict__ bias,
    float* __restrict__ outp, float* __restrict__ hp) {
  __shared__ bf16  xs[kB][kPadD];        // 33.3 KB
  __shared__ float pb[2][kB][kPadP];     // 10.2 KB

  const int t    = blockIdx.x;
  const int tid  = threadIdx.x;
  const int lane = tid & 31;
  const int w    = tid >> 5;
  const int hi   = lane >> 4;
  const int r    = lane & 15;

  if (t == 0) {  // h[0] = h0
    for (int i = tid; i < kBD; i += 256) hp[i] = h0[i];
  }

  const float* xt = x + (size_t)t * kBD;
  for (int i = tid; i < kBD; i += 256)
    xs[i >> 10][i & (kD - 1)] = f2bf(xt[i]);
  __syncthreads();

  // Stage A: 8 waves = {proj 0/1} x {4 N-tiles of R}, full K = 1024 each.
  {
    int p    = w >> 2;
    int tile = w & 3;
    const bf16* V = p ? wVz : wVx;     // [64][1024]
    v8f c = {};
#pragma unroll 4
    for (int kk = 0; kk < 32; ++kk) {
      v16bf a = load_a(&xs[0][0], kPadD, r, kk * 32, hi);
      v16bf b = load_b(V, kD, tile * 16 + r, kk * 32, hi);
      c = wmma_bf16(a, b, c);
    }
#pragma unroll
    for (int j = 0; j < 8; ++j)
      pb[p][j + hi * 8][tile * 16 + r] = c[j];
  }
  __syncthreads();

  // Stage B: 128 jobs = {proj} x {64 N-tiles of D}, K = 64 (2 WMMA steps).
  for (int jj = 0; jj < 16; ++jj) {
    int job = jj * 8 + w;
    int p   = job >> 6;
    int nt  = job & 63;
    const bf16* U = p ? wUz : wUx;     // [1024][64]
    v8f c = {};
#pragma unroll
    for (int kc = 0; kc < 2; ++kc) {
      const float* pr = &pb[p][r][kc * 32 + hi * 8];
      v16bf a;
#pragma unroll
      for (int e = 0; e < 8; ++e) { a[e] = f2bf(pr[e]); a[e + 8] = f2bf(pr[e + 16]); }
      v16bf b = load_b(U, kR, nt * 16 + r, kc * 32, hi);
      c = wmma_bf16(a, b, c);
    }
#pragma unroll
    for (int j = 0; j < 8; ++j) {
      int m = j + hi * 8;
      int n = nt * 16 + r;
      if (p == 0) {
        hp[(size_t)(t + 1) * kBD + m * kD + n] = c[j] + bias[n];   // ux + b
      } else {
        float v = c[j];
        outp[(size_t)t * kBD + m * kD + n] = v / (1.0f + expf(-v)); // silu
      }
    }
  }
}

// ---------------------------------------------------------------------------
// Kernel 2: serial recurrence. One persistent workgroup (8 waves) with
// V_h, U_h, h and P all LDS-resident (~300 KB of the 320 KB WGP LDS).
//   per step: P = h @ Vh^T (K split over 2 wave groups),
//             h' = tanh(P @ Uh^T + staged ux), out *= h'  (staged gate)
// ---------------------------------------------------------------------------
__global__ __launch_bounds__(256) void lr_recurrence(
    const bf16* __restrict__ wVh, const bf16* __restrict__ wUh,
    const float* __restrict__ h0,
    float* __restrict__ outp, float* __restrict__ hp) {
  __shared__ bf16  vh[kR][kPadD];        // 133.1 KB  [n over 64][k over 1024]
  __shared__ bf16  uh[kD * kR];          // 131.1 KB  [n over 1024][k over 64]
  __shared__ bf16  hs[kB][kPadD];        //  33.3 KB
  __shared__ float pb[2][kB][kPadP];     //  10.2 KB

  const int tid  = threadIdx.x;
  const int lane = tid & 31;
  const int w    = tid >> 5;
  const int hi   = lane >> 4;
  const int r    = lane & 15;

  for (int i = tid; i < kW; i += 256) vh[i >> 10][i & (kD - 1)] = wVh[i];
  for (int i = tid; i < kW; i += 256) uh[i] = wUh[i];
  for (int i = tid; i < kBD; i += 256) hs[i >> 10][i & (kD - 1)] = f2bf(h0[i]);

  for (int t = 0; t < kT; ++t) {
    __syncthreads();  // hs/pb ready, prior-step readers done

    // Stage A: P = h @ Vh^T ; 8 waves = {4 N-tiles of R} x {2 K-halves}
    {
      int tile = w & 3;
      int kh   = w >> 2;
      v8f c = {};
#pragma unroll 4
      for (int kk = 0; kk < 16; ++kk) {
        int kb = kh * 512 + kk * 32;
        v16bf a = load_a(&hs[0][0], kPadD, r, kb, hi);
        v16bf b = load_b(&vh[0][0], kPadD, tile * 16 + r, kb, hi);
        c = wmma_bf16(a, b, c);
      }
#pragma unroll
      for (int j = 0; j < 8; ++j)
        pb[kh][j + hi * 8][tile * 16 + r] = c[j];
    }
    __syncthreads();

    // Stage B: 64 N-tiles of D over 8 waves; K = 64 (sum the two K-half P's)
    for (int jj = 0; jj < 8; ++jj) {
      int nt = jj * 8 + w;
      v8f c = {};
#pragma unroll
      for (int kc = 0; kc < 2; ++kc) {
        const float* p0 = &pb[0][r][kc * 32 + hi * 8];
        const float* p1 = &pb[1][r][kc * 32 + hi * 8];
        v16bf a;
#pragma unroll
        for (int e = 0; e < 8; ++e) {
          a[e]     = f2bf(p0[e]      + p1[e]);
          a[e + 8] = f2bf(p0[e + 16] + p1[e + 16]);
        }
        v16bf b = load_b(uh, kR, nt * 16 + r, kc * 32, hi);
        c = wmma_bf16(a, b, c);
      }
#pragma unroll
      for (int j = 0; j < 8; ++j) {
        int m = j + hi * 8;
        int n = nt * 16 + r;
        size_t hoff = (size_t)(t + 1) * kBD + m * kD + n;
        float hv = tanhf(c[j] + hp[hoff]);   // staged value = ux + b
        hp[hoff] = hv;
        size_t ooff = (size_t)t * kBD + m * kD + n;
        outp[ooff] = hv * outp[ooff];        // staged value = silu(uz)
        hs[m][n] = f2bf(hv);
      }
    }
  }
}

// ---------------------------------------------------------------------------
extern "C" void kernel_launch(void* const* d_in, const int* in_sizes, int n_in,
                              void* d_out, int out_size, void* d_ws, size_t ws_size,
                              hipStream_t stream) {
  const float* x  = (const float*)d_in[0];
  const float* h0 = (const float*)d_in[1];
  const float* Uh = (const float*)d_in[2];
  const float* Vh = (const float*)d_in[3];
  const float* Ux = (const float*)d_in[4];
  const float* Vx = (const float*)d_in[5];
  const float* Uz = (const float*)d_in[6];
  const float* Vz = (const float*)d_in[7];
  const float* b  = (const float*)d_in[8];

  float* outp = (float*)d_out;                  // [T,B,D]
  float* hp   = outp + (size_t)kT * kBD;        // [T+1,B,D]
  bf16*  ws   = (bf16*)d_ws;                    // 6 x 64K bf16 = 768 KB

  cvt_weights<<<(6 * kW + 255) / 256, 256, 0, stream>>>(Vh, Uh, Vx, Ux, Vz, Uz, ws);
  lr_input_proj<<<kT, 256, 0, stream>>>(x, h0,
                                        ws + 2 * kW, ws + 3 * kW,   // Vx, Ux
                                        ws + 4 * kW, ws + 5 * kW,   // Vz, Uz
                                        b, outp, hp);
  lr_recurrence<<<1, 256, 0, stream>>>(ws, ws + kW, h0, outp, hp);
}
